// CustomLSTMLayer_2035814499085
// MI455X (gfx1250) — matrix operand order repaired and checked by reading, not compiled, not run on hardware
//
#include <hip/hip_runtime.h>
#include <hip/hip_bf16.h>

// ---------------------------------------------------------------------------
// Problem constants (B=4096, I=H=1024)
// ---------------------------------------------------------------------------
#define B_DIM 4096
#define I_DIM 1024
#define H_DIM 1024
#define G4H   (4 * H_DIM)
#define BH    (B_DIM * H_DIM)

typedef __attribute__((ext_vector_type(16))) __bf16 v16bf;
typedef __attribute__((ext_vector_type(8)))  float  v8f;

union Frag {
    v16bf v;
    uint4 q[2];
};

__device__ __forceinline__ unsigned short f2bf(float f) {
    unsigned int u = __float_as_uint(f);
    unsigned int r = u + 0x7FFFu + ((u >> 16) & 1u);   // round-to-nearest-even
    return (unsigned short)(r >> 16);
}

__device__ __forceinline__ float sigf(float x) {
    return 1.0f / (1.0f + __expf(-x));
}
__device__ __forceinline__ float tanhfastf(float x) {
    return 2.0f * sigf(2.0f * x) - 1.0f;
}

// ---------------------------------------------------------------------------
// Prep kernels
// ---------------------------------------------------------------------------
__global__ __launch_bounds__(256) void k_f32_to_bf16(const float* __restrict__ src,
                                                     unsigned short* __restrict__ dst,
                                                     int n) {
    for (int i = blockIdx.x * blockDim.x + threadIdx.x; i < n;
         i += gridDim.x * blockDim.x) {
        dst[i] = f2bf(src[i]);
    }
}

// Pack 4 gate matrices [H,K] fp32 into one gate-major [4H,K] bf16 buffer.
__global__ __launch_bounds__(256) void k_pack_gates(const float* __restrict__ w0,
                                                    const float* __restrict__ w1,
                                                    const float* __restrict__ w2,
                                                    const float* __restrict__ w3,
                                                    unsigned short* __restrict__ dst,
                                                    int per_gate) {
    const int total = per_gate * 4;
    for (int i = blockIdx.x * blockDim.x + threadIdx.x; i < total;
         i += gridDim.x * blockDim.x) {
        const int g = i / per_gate;
        const int r = i - g * per_gate;
        const float* s = (g == 0) ? w0 : (g == 1) ? w1 : (g == 2) ? w2 : w3;
        dst[i] = f2bf(s[r]);
    }
}

// bias[4H] = bW + bV + b  (gate-major)
__global__ __launch_bounds__(256) void k_bias(const float* __restrict__ bW0, const float* __restrict__ bV0, const float* __restrict__ bb0,
                                              const float* __restrict__ bW1, const float* __restrict__ bV1, const float* __restrict__ bb1,
                                              const float* __restrict__ bW2, const float* __restrict__ bV2, const float* __restrict__ bb2,
                                              const float* __restrict__ bW3, const float* __restrict__ bV3, const float* __restrict__ bb3,
                                              float* __restrict__ bias) {
    int j = blockIdx.x * blockDim.x + threadIdx.x;
    if (j >= G4H) return;
    const int g = j / H_DIM;
    const int r = j - g * H_DIM;
    const float* bW = (g == 0) ? bW0 : (g == 1) ? bW1 : (g == 2) ? bW2 : bW3;
    const float* bV = (g == 0) ? bV0 : (g == 1) ? bV1 : (g == 2) ? bV2 : bV3;
    const float* bb = (g == 0) ? bb0 : (g == 1) ? bb1 : (g == 2) ? bb2 : bb3;
    bias[j] = bW[r] + bV[r] + bb[r];
}

// Per-row L2 norm of x -> active mask (norm > 0.001)
__global__ __launch_bounds__(256) void k_row_mask(const float* __restrict__ x,
                                                  float* __restrict__ mask, int n) {
    __shared__ float red[8];
    const int row = blockIdx.x;
    const float* xr = x + (size_t)row * n;
    float s = 0.0f;
    for (int i = threadIdx.x; i < n; i += blockDim.x) {
        float v = xr[i];
        s += v * v;
    }
    for (int off = 16; off > 0; off >>= 1) s += __shfl_down(s, off, 32);
    const int wid = threadIdx.x >> 5;
    const int ln  = threadIdx.x & 31;
    if (ln == 0) red[wid] = s;
    __syncthreads();
    if (threadIdx.x == 0) {
        float t = 0.0f;
        #pragma unroll
        for (int i = 0; i < 8; ++i) t += red[i];
        mask[row] = (sqrtf(t) > 0.001f) ? 1.0f : 0.0f;
    }
}

// ---------------------------------------------------------------------------
// Fused LSTM GEMM: gates = x@Wx^T + h@Wh^T + bias, then gate nonlinearity.
// Block = 256 threads = 8 waves. Block tile: M=256, N=32 per gate.
// Each wave: M=32 (2 A-frags) x N=32 (2 subtiles) x 4 gates
//   -> 16 v_wmma_f32_16x16x32_bf16 per K-step, 20 b128 loads (1.25 ld/wmma).
// Every B (weight) fragment is reused for two WMMAs; B addresses are
// identical across the 8 waves, so WGP$/L2 absorb the weight traffic.
// ---------------------------------------------------------------------------
__global__ __launch_bounds__(256) void k_lstm_wmma(
    const unsigned short* __restrict__ xb,   // [B, I]  bf16
    const unsigned short* __restrict__ hb,   // [B, H]  bf16
    const unsigned short* __restrict__ wxb,  // [4H, I] bf16 gate-major
    const unsigned short* __restrict__ whb,  // [4H, H] bf16 gate-major
    const float* __restrict__ bias,          // [4H]
    const float* __restrict__ rowmask,       // [B]
    const float* __restrict__ c_prev,        // [B, H]
    float* __restrict__ out)                 // h_next | c_next | c_tilde
{
    constexpr int K = 1024;
    const int lane = threadIdx.x & 31;
    const int wave = threadIdx.x >> 5;
    const int l  = lane & 15;
    const int hi = lane >> 4;
    const int m0 = blockIdx.x * 256 + wave * 32;   // wave owns rows [m0, m0+32)
    const int n0 = blockIdx.y * 32;

    v8f acc[4][2][2];   // [gate][n-subtile][m-subtile]
    {
        v8f z = {};
        #pragma unroll
        for (int g = 0; g < 4; ++g)
            #pragma unroll
            for (int ns = 0; ns < 2; ++ns) {
                acc[g][ns][0] = z;
                acc[g][ns][1] = z;
            }
    }

    #pragma unroll
    for (int src = 0; src < 2; ++src) {
        const unsigned short* A = src ? hb : xb;
        const unsigned short* W = src ? whb : wxb;

        // A fragment (16x32 bf16, ISA A layout): lane l holds row; halves at
        // k0 + hi*8 + {0..7} and k0 + 16 + hi*8 + {0..7}
        const unsigned short* aRow[2];
        #pragma unroll
        for (int ms = 0; ms < 2; ++ms)
            aRow[ms] = A + (size_t)(m0 + ms * 16 + l) * K + hi * 8;

        // B fragment (32x16 bf16, ISA B layout): lane l holds column
        // (= weight row); halves at k0 + hi*8 + {0..7} and k0 + 16 + hi*8 + {0..7}
        const unsigned short* bRow[4][2];
        #pragma unroll
        for (int g = 0; g < 4; ++g)
            #pragma unroll
            for (int ns = 0; ns < 2; ++ns)
                bRow[g][ns] = W + (size_t)(g * H_DIM + n0 + ns * 16 + l) * K + hi * 8;

        #pragma unroll 4
        for (int k0 = 0; k0 < K; k0 += 32) {
            Frag a0, a1;
            {
                const uint4* ap0 = (const uint4*)(aRow[0] + k0);
                a0.q[0] = ap0[0];
                a0.q[1] = ap0[2];
                const uint4* ap1 = (const uint4*)(aRow[1] + k0);
                a1.q[0] = ap1[0];
                a1.q[1] = ap1[2];
            }
            #pragma unroll
            for (int g = 0; g < 4; ++g) {
                #pragma unroll
                for (int ns = 0; ns < 2; ++ns) {
                    Frag b;
                    const uint4* bp = (const uint4*)(bRow[g][ns] + k0);
                    b.q[0] = bp[0];
                    b.q[1] = bp[2];
                    acc[g][ns][0] = __builtin_amdgcn_wmma_f32_16x16x32_bf16(
                        false, a0.v, false, b.v, (short)0, acc[g][ns][0], false, false);
                    acc[g][ns][1] = __builtin_amdgcn_wmma_f32_16x16x32_bf16(
                        false, a1.v, false, b.v, (short)0, acc[g][ns][1], false, false);
                }
            }
        }
    }

    // ---- fused epilogue --------------------------------------------------
    float bcol[4][2];
    #pragma unroll
    for (int g = 0; g < 4; ++g) {
        #pragma unroll
        for (int ns = 0; ns < 2; ++ns) {
            bcol[g][ns] = bias[g * H_DIM + n0 + ns * 16 + l];
        }
    }

    #pragma unroll
    for (int ms = 0; ms < 2; ++ms) {
        #pragma unroll
        for (int v = 0; v < 8; ++v) {
            const int row = m0 + ms * 16 + hi * 8 + v;   // C layout: lanes 16-31 -> M+8
            const float mk = rowmask[row];
            #pragma unroll
            for (int ns = 0; ns < 2; ++ns) {
                const int col = n0 + ns * 16 + l;
                const size_t o = (size_t)row * H_DIM + col;
                const float fp = acc[0][ns][ms][v] + bcol[0][ns];
                const float ip = acc[1][ns][ms][v] + bcol[1][ns];
                const float op = acc[2][ns][ms][v] + bcol[2][ns];
                const float cp = acc[3][ns][ms][v] + bcol[3][ns];
                const float f_t = sigf(fp);
                const float i_t = sigf(ip);
                const float o_t = sigf(op);
                const float ctl = tanhfastf(cp);
                const float cpv = c_prev[o];
                const float c_next = f_t * cpv + i_t * cpv + mk * (i_t * ctl);
                const float h_next = o_t * tanhfastf(c_next);
                out[o]                   = h_next;
                out[(size_t)BH + o]      = c_next;
                out[(size_t)2 * BH + o]  = ctl;
            }
        }
    }
}

// ---------------------------------------------------------------------------
// Host launcher
// ---------------------------------------------------------------------------
extern "C" void kernel_launch(void* const* d_in, const int* in_sizes, int n_in,
                              void* d_out, int out_size, void* d_ws, size_t ws_size,
                              hipStream_t stream) {
    (void)in_sizes; (void)n_in; (void)out_size; (void)ws_size;

    const float* x      = (const float*)d_in[0];
    const float* h_prev = (const float*)d_in[1];
    const float* c_prev = (const float*)d_in[2];
    // d_in[3] = c_prev_tilde_dummy (unused)
    const float* Wf = (const float*)d_in[4];
    const float* bWf = (const float*)d_in[5];
    const float* Vf = (const float*)d_in[6];
    const float* bVf = (const float*)d_in[7];
    const float* Wi = (const float*)d_in[8];
    const float* bWi = (const float*)d_in[9];
    const float* Vi = (const float*)d_in[10];
    const float* bVi = (const float*)d_in[11];
    const float* Wo = (const float*)d_in[12];
    const float* bWo = (const float*)d_in[13];
    const float* Vo = (const float*)d_in[14];
    const float* bVo = (const float*)d_in[15];
    const float* Wc = (const float*)d_in[16];
    const float* bWc = (const float*)d_in[17];
    const float* Vc = (const float*)d_in[18];
    const float* bVc = (const float*)d_in[19];
    const float* bf_ = (const float*)d_in[20];
    const float* bi_ = (const float*)d_in[21];
    const float* bo_ = (const float*)d_in[22];
    const float* bc_ = (const float*)d_in[23];

    // Workspace layout
    unsigned short* xb  = (unsigned short*)d_ws;              // B*I bf16  (8 MiB)
    unsigned short* hb  = xb  + (size_t)B_DIM * I_DIM;        // B*H bf16  (8 MiB)
    unsigned short* wxb = hb  + (size_t)B_DIM * H_DIM;        // 4H*I bf16 (8 MiB)
    unsigned short* whb = wxb + (size_t)G4H * I_DIM;          // 4H*H bf16 (8 MiB)
    float* bias  = (float*)(whb + (size_t)G4H * H_DIM);       // 4H f32
    float* maskb = bias + G4H;                                // B f32

    float* out = (float*)d_out;

    // ---- prep ----
    k_f32_to_bf16<<<4096, 256, 0, stream>>>(x, xb, B_DIM * I_DIM);
    k_f32_to_bf16<<<4096, 256, 0, stream>>>(h_prev, hb, B_DIM * H_DIM);
    k_pack_gates<<<4096, 256, 0, stream>>>(Wf, Wi, Wo, Wc, wxb, H_DIM * I_DIM);
    k_pack_gates<<<4096, 256, 0, stream>>>(Vf, Vi, Vo, Vc, whb, H_DIM * H_DIM);
    k_bias<<<G4H / 256, 256, 0, stream>>>(bWf, bVf, bf_, bWi, bVi, bi_,
                                          bWo, bVo, bo_, bWc, bVc, bc_, bias);
    k_row_mask<<<B_DIM, 256, 0, stream>>>(x, maskb, I_DIM);

    // ---- fused WMMA GEMM + gate epilogue ----
    dim3 grid(B_DIM / 256, H_DIM / 32);
    k_lstm_wmma<<<grid, 256, 0, stream>>>(xb, hb, wxb, whb, bias, maskb, c_prev, out);
}
